// HGSELLayerFast_40664750359237
// MI455X (gfx1250) — matrix operand order repaired and checked by reading, not compiled, not run on hardware
//
#include <hip/hip_runtime.h>
#include <hip/hip_bf16.h>
#include <math.h>

typedef __attribute__((ext_vector_type(16))) __bf16 v16bf;
typedef __attribute__((ext_vector_type(8)))  __bf16 v8bf;
typedef __attribute__((ext_vector_type(8)))  float  v8f;

namespace {
constexpr int Ttok    = 8192;   // B*S
constexpr int Ddim    = 512;
constexpr int Fdim    = 2048;
constexpr int Eexp    = 64;
constexpr int Ksel    = 2;
constexpr int CAP     = 2048;   // CAP_FACTOR*T*k/E
constexpr int NASSIGN = Ttok * Ksel; // 16384
}

// ---------------------------------------------------------------- routing
__global__ void route_kernel(const float* __restrict__ x,
                             const float* __restrict__ hp,
                             int* __restrict__ sel) {
  __shared__ float xs[Ddim];
  __shared__ float ss[256];
  __shared__ int   si[256];
  const int t   = blockIdx.x;
  const int tid = threadIdx.x;
  xs[tid]       = x[(size_t)t * Ddim + tid];
  xs[tid + 256] = x[(size_t)t * Ddim + tid + 256];
  __syncthreads();
  const int h = tid >> 6;
  const int e = tid & 63;
  const float* w = hp + (size_t)h * Ddim * Eexp + e;
  float acc = 0.f;
#pragma unroll 8
  for (int d = 0; d < Ddim; ++d) acc = fmaf(xs[d], w[(size_t)d * Eexp], acc);
  ss[tid] = acc; si[tid] = e;
  __syncthreads();
  // argmax over 64 experts per hash, lowest index wins ties (matches jnp.argmax)
#pragma unroll
  for (int off = 32; off > 0; off >>= 1) {
    if (e < off) {
      float o = ss[tid + off]; int oi = si[tid + off];
      if (o > ss[tid] || (o == ss[tid] && oi < si[tid])) { ss[tid] = o; si[tid] = oi; }
    }
    __syncthreads();
  }
  if (tid == 0) {
    float sc[4] = { ss[0], ss[64], ss[128], ss[192] };
    int   cd[4] = { si[0], si[64], si[128], si[192] };
    int b0 = 0;
#pragma unroll
    for (int hh = 1; hh < 4; ++hh) if (sc[hh] > sc[b0]) b0 = hh;   // ties -> lower h
    int b1 = -1;
#pragma unroll
    for (int hh = 0; hh < 4; ++hh) {
      if (hh == b0) continue;
      if (b1 < 0 || sc[hh] > sc[b1]) b1 = hh;
    }
    sel[2 * t]     = cd[b0];
    sel[2 * t + 1] = cd[b1];
  }
}

// ------------------------------------------------ token-major positions / capacity
__global__ void count_pos_kernel(const int* __restrict__ sel,
                                 int* __restrict__ pos,
                                 int* __restrict__ counts) {
  const int e   = blockIdx.x;
  const int tid = threadIdx.x;
  constexpr int CH = NASSIGN / 256; // 64
  __shared__ int s[256];
  const int base = tid * CH;
  int cnt = 0;
  for (int j = 0; j < CH; ++j) cnt += (sel[base + j] == e) ? 1 : 0;
  s[tid] = cnt;
  __syncthreads();
  for (int off = 1; off < 256; off <<= 1) {   // inclusive Hillis-Steele scan
    int v = (tid >= off) ? s[tid - off] : 0;
    __syncthreads();
    s[tid] += v;
    __syncthreads();
  }
  int run = s[tid] - cnt;                     // exclusive prefix
  for (int j = 0; j < CH; ++j) {
    if (sel[base + j] == e) pos[base + j] = run++;
  }
  if (tid == 255) {
    int tot = s[255];
    counts[e] = (tot < CAP) ? tot : CAP;      // kept rows
  }
}

__global__ void offsets_kernel(const int* __restrict__ counts, int* __restrict__ offs) {
  if (threadIdx.x == 0) {
    int r = 0;
    for (int e = 0; e < Eexp; ++e) { offs[e] = r; r += counts[e]; }
  }
}

// ---------------------------------------------------------------- dispatch
__global__ void dispatch_kernel(const float* __restrict__ x,
                                const int* __restrict__ sel,
                                const int* __restrict__ pos,
                                const int* __restrict__ offs,
                                int* __restrict__ rowIdx,
                                __bf16* __restrict__ xc) {
  const int i   = blockIdx.x;
  const int tid = threadIdx.x;
  const int e = sel[i];
  const int p = pos[i];
  const bool valid = (p < CAP);
  const int row = offs[e] + p;
  if (tid == 0) rowIdx[i] = valid ? row : -1;
  if (!valid) return;
  const int t = i >> 1;
  const float* src = x + (size_t)t * Ddim;
  __bf16* dst = xc + (size_t)row * Ddim;
  for (int d = tid; d < Ddim; d += 128) dst[d] = (__bf16)src[d];
}

// ------------------------------------------------- WMMA fragment helpers
// A-matrix 16x32 bf16 fragment (contiguous K order): lane L<16 -> row L, K={0..15};
// lane L>=16 -> row L-16, K={16..31}. Two contiguous 16B LDS loads.
__device__ __forceinline__ v16bf load_a_frag(const __bf16* p, int lane) {
  const int kb = (lane >> 4) << 4;           // 0 or 16
  const v8bf lo = *(const v8bf*)(p + kb);
  const v8bf hi = *(const v8bf*)(p + 8 + kb);
  v16bf a;
#pragma unroll
  for (int i = 0; i < 8; ++i) { a[i] = lo[i]; a[i + 8] = hi[i]; }
  return a;
}

// ---------------------------------------------------------------- fused expert FFN
// grid = (CAP/64 tiles, E experts), block = 256 (8 waves)
__global__ void __launch_bounds__(256)
ffn_kernel(const __bf16* __restrict__ xc,
           const float* __restrict__ W1,
           const float* __restrict__ W2,
           const int* __restrict__ offs,
           const int* __restrict__ counts,
           float* __restrict__ yc) {
  const int mt = blockIdx.x;
  const int e  = blockIdx.y;
  const int kept = counts[e];
  if (mt * 64 >= kept) return;
  const int row0   = offs[e] + mt * 64;
  const int rem    = kept - mt * 64;
  const int nvalid = rem < 64 ? rem : 64;

  __shared__ __attribute__((aligned(64))) __bf16 Xs[64 * 512];   // X tile, K-contig
  __shared__ __attribute__((aligned(64))) __bf16 Hs[64 * 128];   // gelu(H) chunk, K-contig
  __shared__ __attribute__((aligned(64))) __bf16 W1s[128 * 32];  // [f][k]  K-contig
  __shared__ __attribute__((aligned(64))) __bf16 W2s[512 * 32];  // [n][k]  K-contig

  const int tid  = threadIdx.x;
  const int w    = tid >> 5;
  const int lane = tid & 31;
  const int col  = lane & 15;
  const int rb   = (lane >> 4) << 3;
  const int m_idx  = w & 3;   // stage-A row tile
  const int n_half = w >> 2;  // stage-A 64-col half
  const int kb16 = (lane >> 4) << 4;

  // load 64 x 512 bf16 X tile (64 KB) with 128-bit copies
  {
    const uint4* src = (const uint4*)(xc + (size_t)row0 * 512);
    uint4* dst = (uint4*)Xs;
#pragma unroll
    for (int i = 0; i < 16; ++i) dst[tid + 256 * i] = src[tid + 256 * i];
  }

  const v8f zero = {};
  v8f accY[4][4];
#pragma unroll
  for (int m = 0; m < 4; ++m)
#pragma unroll
    for (int n = 0; n < 4; ++n) accY[m][n] = zero;

  const int r32  = tid >> 3;        // 0..31 : k row within weight chunk
  const int fb16 = (tid & 7) << 4;  // W1 chunk: 16 consecutive f per thread
  const int nb64 = (tid & 7) << 6;  // W2 chunk: 64 consecutive n per thread

  for (int fc = 0; fc < Fdim; fc += 128) {
    v8f accH[4];
#pragma unroll
    for (int n = 0; n < 4; ++n) accH[n] = zero;

    // ---- stage A: Hc[64,128] = X[64,512] x W1[:, fc:fc+128]
    for (int k0 = 0; k0 < 512; k0 += 32) {
      __syncthreads();
      {   // W1[k0+r][fc+f] fp32 -> bf16, stored transposed [f][k]
        const float* g = W1 + ((size_t)e * Ddim + k0 + r32) * Fdim + fc + fb16;
#pragma unroll
        for (int j = 0; j < 4; ++j) {
          const float4 v = ((const float4*)g)[j];
          const int f = fb16 + j * 4;
          W1s[(f + 0) * 32 + r32] = (__bf16)v.x;
          W1s[(f + 1) * 32 + r32] = (__bf16)v.y;
          W1s[(f + 2) * 32 + r32] = (__bf16)v.z;
          W1s[(f + 3) * 32 + r32] = (__bf16)v.w;
        }
      }
      __syncthreads();
      const v16bf a = load_a_frag(Xs + (m_idx * 16 + col) * 512 + k0, lane);
#pragma unroll
      for (int n = 0; n < 4; ++n) {
        const v16bf b = *(const v16bf*)(W1s + (n_half * 64 + n * 16 + col) * 32 + kb16);
        accH[n] = __builtin_amdgcn_wmma_f32_16x16x32_bf16(false, a, false, b,
                                                          (short)0, accH[n], false, false);
      }
    }
    __syncthreads();
    // exact GELU (approximate=False) then bf16 into Hs
#pragma unroll
    for (int n = 0; n < 4; ++n) {
#pragma unroll
      for (int i = 0; i < 8; ++i) {
        const float v = accH[n][i];
        const float g = 0.5f * v * (1.0f + erff(v * 0.70710678118654752f));
        Hs[(m_idx * 16 + rb + i) * 128 + n_half * 64 + n * 16 + col] = (__bf16)g;
      }
    }

    // ---- stage B: Y[64,512] += Hc[64,128] x W2[fc:fc+128, :]
    for (int kk = 0; kk < 128; kk += 32) {
      __syncthreads();
      {   // W2[fc+kk+r][n] fp32 -> bf16, stored transposed [n][k]
        const float* g2 = W2 + ((size_t)e * Fdim + fc + kk + r32) * Ddim + nb64;
#pragma unroll
        for (int j = 0; j < 16; ++j) {
          const float4 v = ((const float4*)g2)[j];
          const int n = nb64 + j * 4;
          W2s[(n + 0) * 32 + r32] = (__bf16)v.x;
          W2s[(n + 1) * 32 + r32] = (__bf16)v.y;
          W2s[(n + 2) * 32 + r32] = (__bf16)v.z;
          W2s[(n + 3) * 32 + r32] = (__bf16)v.w;
        }
      }
      __syncthreads();
#pragma unroll
      for (int m = 0; m < 4; ++m) {
        const v16bf a = load_a_frag(Hs + (m * 16 + col) * 128 + kk, lane);
#pragma unroll
        for (int n = 0; n < 4; ++n) {
          const v16bf b = *(const v16bf*)(W2s + (w * 64 + n * 16 + col) * 32 + kb16);
          accY[m][n] = __builtin_amdgcn_wmma_f32_16x16x32_bf16(false, a, false, b,
                                                               (short)0, accY[m][n], false, false);
        }
      }
    }
  }

  // store Y rows (mask padded rows)
#pragma unroll
  for (int m = 0; m < 4; ++m) {
#pragma unroll
    for (int i = 0; i < 8; ++i) {
      const int r = m * 16 + rb + i;
      if (r < nvalid) {
        float* o = yc + (size_t)(row0 + r) * Ddim + w * 64 + col;
#pragma unroll
        for (int n = 0; n < 4; ++n) o[n * 16] = accY[m][n][i];
      }
    }
  }
}

// ---------------------------------------------------------------- combine (mean over k)
__global__ void combine_kernel(const float* __restrict__ yc,
                               const int* __restrict__ rowIdx,
                               float* __restrict__ out) {
  const int t   = blockIdx.x;
  const int tid = threadIdx.x;
  const int r0 = rowIdx[2 * t];
  const int r1 = rowIdx[2 * t + 1];
#pragma unroll
  for (int d = tid; d < Ddim; d += 256) {
    const float a = (r0 >= 0) ? yc[(size_t)r0 * Ddim + d] : 0.f;
    const float b = (r1 >= 0) ? yc[(size_t)r1 * Ddim + d] : 0.f;
    out[(size_t)t * Ddim + d] = 0.5f * (a + b);
  }
}

extern "C" void kernel_launch(void* const* d_in, const int* in_sizes, int n_in,
                              void* d_out, int out_size, void* d_ws, size_t ws_size,
                              hipStream_t stream) {
  (void)in_sizes; (void)n_in; (void)out_size; (void)ws_size;
  const float* x  = (const float*)d_in[0];   // [T, D] fp32
  const float* hp = (const float*)d_in[1];   // [H, D, E] fp32
  const float* W1 = (const float*)d_in[2];   // [E, D, F] fp32
  const float* W2 = (const float*)d_in[3];   // [E, F, D] fp32
  float* out = (float*)d_out;

  char* ws = (char*)d_ws;
  size_t off = 0;
  auto take = [&](size_t bytes) -> void* {
    void* p = ws + off;
    off = (off + bytes + 255) & ~(size_t)255;
    return p;
  };
  int*    sel    = (int*)take(sizeof(int) * NASSIGN);
  int*    pos    = (int*)take(sizeof(int) * NASSIGN);
  int*    counts = (int*)take(sizeof(int) * Eexp);
  int*    offs   = (int*)take(sizeof(int) * Eexp);
  int*    rowIdx = (int*)take(sizeof(int) * NASSIGN);
  __bf16* xc     = (__bf16*)take(sizeof(__bf16) * (size_t)(NASSIGN + 64) * Ddim);
  float*  yc     = (float*)take(sizeof(float) * (size_t)(NASSIGN + 64) * Ddim);

  route_kernel    <<<Ttok,    256, 0, stream>>>(x, hp, sel);
  count_pos_kernel<<<Eexp,    256, 0, stream>>>(sel, pos, counts);
  offsets_kernel  <<<1,        64, 0, stream>>>(counts, offs);
  dispatch_kernel <<<NASSIGN, 128, 0, stream>>>(x, sel, pos, offs, rowIdx, xc);
  ffn_kernel      <<<dim3(CAP / 64, Eexp), 256, 0, stream>>>(xc, W1, W2, offs, counts, yc);
  combine_kernel  <<<Ttok,    256, 0, stream>>>(yc, rowIdx, out);
}
